// GraphAttentionLayer_1039382085705
// MI455X (gfx1250) — matrix operand order repaired and checked by reading, not compiled, not run on hardware
//
#include <hip/hip_runtime.h>
#include <hip/hip_bf16.h>

// GAT: B=8, N=2048, F_in=F_out=64
#define GAT_B 8
#define GAT_N 2048
#define GAT_F 64
#define GAT_ALPHA 0.2f
#define GAT_NEGINF (-9.0e15f)

typedef __attribute__((ext_vector_type(16))) _Float16 v16h;
typedef __attribute__((ext_vector_type(8)))  _Float16 v8h;
typedef __attribute__((ext_vector_type(8)))  float    v8f;

// -------------------------------------------------------------------------
// Kernel 1: Wh = h @ W  (per (b,n) row), store WhT in f16 (B-operand layout
// feeder: WhT[b][f][n], contiguous along n so per-lane K runs are contiguous
// after the transpose), plus s1 = Wh . a1, s2 = Wh . a2 in f32.
// -------------------------------------------------------------------------
__global__ __launch_bounds__(64)
void gat_wh_kernel(const float* __restrict__ h,
                   const float* __restrict__ W,
                   const float* __restrict__ a,
                   _Float16* __restrict__ WhT,
                   float* __restrict__ s1g,
                   float* __restrict__ s2g) {
    const int n = blockIdx.x;      // row within batch
    const int b = blockIdx.y;      // batch
    const int t = threadIdx.x;     // output feature 0..63

    __shared__ float hrow[GAT_F];
    __shared__ float whs[GAT_F];

    hrow[t] = h[((size_t)b * GAT_N + n) * GAT_F + t];
    __syncthreads();

    float acc = 0.0f;
    #pragma unroll 8
    for (int k = 0; k < GAT_F; ++k) {
        acc = fmaf(hrow[k], W[k * GAT_F + t], acc);
    }
    whs[t] = acc;
    // transposed f16 copy: WhT[(b*64 + t)*2048 + n]
    WhT[((size_t)b * GAT_F + t) * GAT_N + n] = (_Float16)acc;
    __syncthreads();

    if (t == 0) {
        float s = 0.0f;
        #pragma unroll 8
        for (int k = 0; k < GAT_F; ++k) s = fmaf(whs[k], a[k], s);
        s1g[(size_t)b * GAT_N + n] = s;
    } else if (t == 1) {
        float s = 0.0f;
        #pragma unroll 8
        for (int k = 0; k < GAT_F; ++k) s = fmaf(whs[k], a[GAT_F + k], s);
        s2g[(size_t)b * GAT_N + n] = s;
    }
}

// -------------------------------------------------------------------------
// Kernel 2: fused masked-softmax + (attn @ Wh), one block (64 thr = 2 waves)
// per (batch, 16-row tile).
//   Pass A: online (max, sum-exp) per row across all 2048 neighbors.
//   Pass B: per 32-wide K chunk, build P tile in WMMA A layout, two
//           v_wmma_f32_16x16x32_f16 per wave (wave owns 32 output cols).
// -------------------------------------------------------------------------
__global__ __launch_bounds__(64)
void gat_attn_kernel(const int* __restrict__ adj,
                     const _Float16* __restrict__ WhT,
                     const float* __restrict__ s1g,
                     const float* __restrict__ s2g,
                     float* __restrict__ out) {
    const int tile = blockIdx.x;          // 0..127
    const int bb   = blockIdx.y;          // batch
    const int tid  = threadIdx.x;         // 0..63
    const int i0   = tile * 16;

    __shared__ float s2s[GAT_N];          // 8 KB: s2 for this batch
    __shared__ float s1s[16];
    __shared__ float pmx[64];
    __shared__ float psum[64];
    __shared__ float mxr[16];
    __shared__ float invr[16];

    // ---- preload s2 (2048 floats, float4 per step) and s1 tile ----
    {
        const float4* s2v = (const float4*)(s2g + (size_t)bb * GAT_N);
        float4* dst = (float4*)s2s;
        #pragma unroll
        for (int m = 0; m < 8; ++m) dst[tid + 64 * m] = s2v[tid + 64 * m];
        if (tid < 16) s1s[tid] = s1g[(size_t)bb * GAT_N + i0 + tid];
    }
    __syncthreads();

    // ---- Pass A: per-row online max / sum(exp) ----
    {
        const int row  = tid >> 2;        // 0..15
        const int part = tid & 3;         // 0..3
        const float s1i = s1s[row];
        const int* arow = adj + (size_t)(i0 + row) * GAT_N;
        float mx = GAT_NEGINF;
        float sum = 0.0f;
        for (int m = 0; m < 128; ++m) {
            const int j0 = part * 4 + 16 * m;
            const int4 av4 = *(const int4*)(arow + j0);
            #pragma unroll
            for (int c = 0; c < 4; ++c) {
                const int ac = (&av4.x)[c];
                float sc = s1i + s2s[j0 + c];
                sc = sc > 0.0f ? sc : GAT_ALPHA * sc;
                sc = ac > 0 ? sc : GAT_NEGINF;
                const float nm = fmaxf(mx, sc);
                sum = sum * __expf(mx - nm) + __expf(sc - nm);
                mx = nm;
            }
        }
        pmx[tid] = mx;
        psum[tid] = sum;
    }
    __syncthreads();
    if ((tid & 3) == 0) {
        float mx = pmx[tid], sum = psum[tid];
        #pragma unroll
        for (int c = 1; c < 4; ++c) {
            const float m2 = pmx[tid + c], sm2 = psum[tid + c];
            const float nm = fmaxf(mx, m2);
            sum = sum * __expf(mx - nm) + sm2 * __expf(m2 - nm);
            mx = nm;
        }
        mxr[tid >> 2] = mx;
        invr[tid >> 2] = 1.0f / sum;
    }
    __syncthreads();

    // ---- Pass B: WMMA over K chunks of 32 ----
    const int lane = tid & 31;
    const int w    = tid >> 5;            // wave: 0 -> cols 0..31, 1 -> 32..63
    const int row  = lane & 15;           // A/C row group
    const int loff = (lane < 16) ? 0 : 8; // A-operand K lane offset
    const int kb   = (lane < 16) ? 0 : 8; // B-operand K lane offset
    const float mxi = mxr[row];
    const float invi = invr[row];
    const float s1i = s1s[row];
    const int* arow = adj + (size_t)(i0 + row) * GAT_N;
    const int ncol = w * 32 + (lane & 15);
    const _Float16* bptr0 = WhT + ((size_t)bb * GAT_F + ncol) * GAT_N + kb;
    const _Float16* bptr1 = WhT + ((size_t)bb * GAT_F + ncol + 16) * GAT_N + kb;

    v8f c0 = {};
    v8f c1 = {};

    for (int ch = 0; ch < 64; ++ch) {
        const int j0 = ch * 32;

        // adjacency for this lane's 16 K slots (two contiguous runs of 8)
        int av[16];
        *(int4*)&av[0]  = *(const int4*)(arow + j0 + loff);
        *(int4*)&av[4]  = *(const int4*)(arow + j0 + loff + 4);
        *(int4*)&av[8]  = *(const int4*)(arow + j0 + loff + 16);
        *(int4*)&av[12] = *(const int4*)(arow + j0 + loff + 20);

        // A operand: probabilities in the native 16-bit A-matrix layout.
        // element e<8  -> K = loff + e ; e>=8 -> K = loff + 16 + (e-8)
        v16h amat;
        #pragma unroll
        for (int e = 0; e < 16; ++e) {
            const int k = (e < 8) ? (loff + e) : (loff + 8 + e);
            float sc = s1i + s2s[j0 + k];
            sc = sc > 0.0f ? sc : GAT_ALPHA * sc;
            sc = av[e] > 0 ? sc : GAT_NEGINF;
            amat[e] = (_Float16)(__expf(sc - mxi) * invi);
        }

        // B operands from WhT (K-contiguous per lane): two b128 loads each
        v16h b0, b1;
        {
            const v8h lo = *(const v8h*)(bptr0 + j0);
            const v8h hi = *(const v8h*)(bptr0 + j0 + 16);
            #pragma unroll
            for (int e = 0; e < 8; ++e) { b0[e] = lo[e]; b0[e + 8] = hi[e]; }
        }
        {
            const v8h lo = *(const v8h*)(bptr1 + j0);
            const v8h hi = *(const v8h*)(bptr1 + j0 + 16);
            #pragma unroll
            for (int e = 0; e < 8; ++e) { b1[e] = lo[e]; b1[e + 8] = hi[e]; }
        }

        c0 = __builtin_amdgcn_wmma_f32_16x16x32_f16(false, amat, false, b0,
                                                    (short)0, c0, false, false);
        c1 = __builtin_amdgcn_wmma_f32_16x16x32_f16(false, amat, false, b1,
                                                    (short)0, c1, false, false);
    }

    // ---- store D tile: C layout -> row = r + (lane<16?0:8), col = lane&15 ----
    float* orow = out + ((size_t)bb * GAT_N + i0) * GAT_F;
    const int mo = (lane < 16) ? 0 : 8;
    #pragma unroll
    for (int r = 0; r < 8; ++r) {
        orow[(size_t)(r + mo) * GAT_F + ncol]      = c0[r];
        orow[(size_t)(r + mo) * GAT_F + ncol + 16] = c1[r];
    }
}

// -------------------------------------------------------------------------
// Host launcher
// -------------------------------------------------------------------------
extern "C" void kernel_launch(void* const* d_in, const int* in_sizes, int n_in,
                              void* d_out, int out_size, void* d_ws, size_t ws_size,
                              hipStream_t stream) {
    const float* h   = (const float*)d_in[0];  // (8,2048,64) f32
    const int*   adj = (const int*)  d_in[1];  // (2048,2048) i32
    const float* W   = (const float*)d_in[2];  // (64,64) f32
    const float* a   = (const float*)d_in[3];  // (128,) f32
    float* out = (float*)d_out;                // (8,2048,64) f32

    char* ws = (char*)d_ws;
    _Float16* WhT = (_Float16*)ws;                         // 8*64*2048 f16 = 2 MB
    float* s1g = (float*)(ws + (size_t)2 * 1024 * 1024);   // 8*2048 f32
    float* s2g = s1g + (size_t)GAT_B * GAT_N;              // 8*2048 f32

    {
        dim3 grid(GAT_N, GAT_B);
        gat_wh_kernel<<<grid, 64, 0, stream>>>(h, W, a, WhT, s1g, s2g);
    }
    {
        dim3 grid(GAT_N / 16, GAT_B);
        gat_attn_kernel<<<grid, 64, 0, stream>>>(adj, WhT, s1g, s2g, out);
    }
}
